// MultivariatePosEncTransformer_49692771615224
// MI455X (gfx1250) — matrix operand order repaired and checked by reading, not compiled, not run on hardware
//
#include <hip/hip_runtime.h>

// ---------------- problem constants ----------------
#define BB 16
#define SS 512
#define INF_ 8
#define DD 512
#define HH 8
#define LL 4
#define FF_ 2048
#define HD 64
#define TOK (BB * SS)          // 8192

typedef __bf16 bf16_t;
typedef __attribute__((ext_vector_type(16))) __bf16 v16bf;
typedef __attribute__((ext_vector_type(8)))  float  v8f;
typedef __attribute__((ext_vector_type(8)))  unsigned int v8u;

// ---------------- workspace layout (bytes) ----------------
static const size_t OFF_X   = 0;
static const size_t OFF_XB  = OFF_X  + (size_t)TOK * DD * 4;
static const size_t OFF_Y   = OFF_XB + (size_t)TOK * DD * 2;
static const size_t OFF_SH  = OFF_Y  + (size_t)TOK * DD * 4;
static const size_t OFF_QKV = OFF_SH;
static const size_t OFF_VT  = OFF_QKV + (size_t)TOK * 3 * DD * 2;
static const size_t OFF_CTX = OFF_VT  + (size_t)BB * HH * HD * SS * 2;
static const size_t OFF_H   = OFF_SH; // reused for FFN hidden
static const size_t SH_BYTES = (size_t)TOK * 3 * DD * 2
                             + (size_t)BB * HH * HD * SS * 2
                             + (size_t)TOK * DD * 2;
static const size_t OFF_W   = OFF_SH + SH_BYTES;
static const size_t PW_QKV = (size_t)DD * 3 * DD;
static const size_t PW_OW  = (size_t)DD * DD;
static const size_t PW_F1  = (size_t)DD * FF_;
static const size_t PW_F2  = (size_t)FF_ * DD;
static const size_t PW_LAYER = PW_QKV + PW_OW + PW_F1 + PW_F2;

// ---------------- WMMA helper ----------------
__device__ __forceinline__ v8f bwmma(v16bf a, v16bf b, v8f c) {
  return __builtin_amdgcn_wmma_f32_16x16x32_bf16(false, a, false, b, (short)0, c,
                                                 false, false);
}

// A fragment (16-bit A 16x32, contiguous k order): lane (m=lane&15, hf=lane>>4)
// holds K = kbase+16*hf+0..15 of row m.
__device__ __forceinline__ v16bf load_a2(const bf16_t* p0, const bf16_t* p1) {
  uint4 a = *(const uint4*)p0;
  uint4 b = *(const uint4*)p1;
  v8u u = {a.x, a.y, a.z, a.w, b.x, b.y, b.z, b.w};
  return __builtin_bit_cast(v16bf, u);
}

// ---------------- VALU cross-lane reductions (no LDS) ----------------
__device__ __forceinline__ float pl16(float v, unsigned lo, unsigned hi) {
  int s = __float_as_int(v);
  int p = __builtin_amdgcn_permlane16(s, s, lo, hi, false, false);
  return __int_as_float(p);
}
__device__ __forceinline__ float plx16(float v) { // xor 16 (other row, same slot)
  int s = __float_as_int(v);
  int p = __builtin_amdgcn_permlanex16(s, s, 0x76543210u, 0xFEDCBA98u, false, false);
  return __int_as_float(p);
}
__device__ __forceinline__ float redmax16(float v) {
  v = fmaxf(v, pl16(v, 0x67452301u, 0xEFCDAB89u)); // xor 1
  v = fmaxf(v, pl16(v, 0x54761032u, 0xDCFE98BAu)); // xor 2
  v = fmaxf(v, pl16(v, 0x32107654u, 0xBA98FEDCu)); // xor 4
  v = fmaxf(v, pl16(v, 0xFEDCBA98u, 0x76543210u)); // xor 8
  return v;
}
__device__ __forceinline__ float redsum16(float v) {
  v += pl16(v, 0x67452301u, 0xEFCDAB89u);
  v += pl16(v, 0x54761032u, 0xDCFE98BAu);
  v += pl16(v, 0x32107654u, 0xBA98FEDCu);
  v += pl16(v, 0xFEDCBA98u, 0x76543210u);
  return v;
}
__device__ __forceinline__ float redsum32(float v) {
  v = redsum16(v);
  v += plx16(v);
  return v;
}

// ---------------- weight packing into B-fragment layout ----------------
// B fragment, contiguous k order: lane (n=lane&15, hf=lane>>4) holds K = 16*hf+0..15
// for column n. Packed: [(nt16*kcCount + kc)*32 + lane]*16 + e (32B/lane).
__global__ void pack_b_k(const float* __restrict__ W, bf16_t* __restrict__ P,
                         int K, int N) {
  int idx = blockIdx.x * 256 + threadIdx.x;
  if (idx >= K * N) return;
  int e     = idx & 15;
  int lane  = (idx >> 4) & 31;
  int chunk = idx >> 9;
  int kcCount = K >> 5;
  int nt = chunk / kcCount;
  int kc = chunk - nt * kcCount;
  int n  = nt * 16 + (lane & 15);
  int k  = kc * 32 + (lane >> 4) * 16 + e;
  P[idx] = (bf16_t)W[(size_t)k * N + n];
}

// ---------------- generic WMMA GEMM, 32x32 output tile per wave ----------------
// 4 WMMAs per K=32 chunk on 2 A + 2 B fragments: 2 b128 loads per WMMA.
// MODE 0: f32 out; 1: bf16 out; 2: bf16 + ReLU; 3: bf16, cols<512 scaled 0.125
template <int MODE>
__global__ void wmma_gemm_k(const bf16_t* __restrict__ A,
                            const bf16_t* __restrict__ Bp,
                            const float* __restrict__ bias,
                            void* __restrict__ out, int M, int N, int K) {
  int wave = threadIdx.x >> 5;
  int lane = threadIdx.x & 31;
  int tile = blockIdx.x * 4 + wave;          // 32x32 tiles
  int ntiles = N >> 5;
  int mt = tile / ntiles;
  int nt = tile - mt * ntiles;
  int hf = lane >> 4;
  int lq = lane & 15;
  int kcCount = K >> 5;
  const bf16_t* arow0 = A + (size_t)(mt * 32 + lq) * K;
  const bf16_t* arow1 = arow0 + (size_t)16 * K;
  const bf16_t* bb0   = Bp + (size_t)(nt * 2) * kcCount * 512 + lane * 16;
  const bf16_t* bb1   = bb0 + (size_t)kcCount * 512;
  v8f c00 = {}, c01 = {}, c10 = {}, c11 = {};
#pragma unroll 2
  for (int kc = 0; kc < kcCount; ++kc) {
    v16bf a0 = load_a2(arow0 + kc * 32 + 16 * hf, arow0 + kc * 32 + 16 * hf + 8);
    v16bf a1 = load_a2(arow1 + kc * 32 + 16 * hf, arow1 + kc * 32 + 16 * hf + 8);
    v16bf b0 = *(const v16bf*)(bb0 + kc * 512);
    v16bf b1 = *(const v16bf*)(bb1 + kc * 512);
    c00 = bwmma(a0, b0, c00);
    c01 = bwmma(a0, b1, c01);
    c10 = bwmma(a1, b0, c10);
    c11 = bwmma(a1, b1, c11);
  }
  int n0 = nt * 32 + lq;
  int n1 = n0 + 16;
  float bi0 = bias[n0], bi1 = bias[n1];
  float sc0 = (MODE == 3 && n0 < DD) ? 0.125f : 1.f;
  float sc1 = (MODE == 3 && n1 < DD) ? 0.125f : 1.f;
#pragma unroll
  for (int r = 0; r < 8; ++r) {
    size_t row0 = (size_t)(mt * 32 + r + 8 * hf);
    size_t row1 = row0 + 16;
    float v00 = (c00[r] + bi0) * sc0;
    float v01 = (c01[r] + bi1) * sc1;
    float v10 = (c10[r] + bi0) * sc0;
    float v11 = (c11[r] + bi1) * sc1;
    if (MODE == 0) {
      ((float*)out)[row0 * N + n0] = v00;
      ((float*)out)[row0 * N + n1] = v01;
      ((float*)out)[row1 * N + n0] = v10;
      ((float*)out)[row1 * N + n1] = v11;
    } else if (MODE == 2) {
      ((bf16_t*)out)[row0 * N + n0] = (bf16_t)fmaxf(v00, 0.f);
      ((bf16_t*)out)[row0 * N + n1] = (bf16_t)fmaxf(v01, 0.f);
      ((bf16_t*)out)[row1 * N + n0] = (bf16_t)fmaxf(v10, 0.f);
      ((bf16_t*)out)[row1 * N + n1] = (bf16_t)fmaxf(v11, 0.f);
    } else {
      ((bf16_t*)out)[row0 * N + n0] = (bf16_t)v00;
      ((bf16_t*)out)[row0 * N + n1] = (bf16_t)v01;
      ((bf16_t*)out)[row1 * N + n0] = (bf16_t)v10;
      ((bf16_t*)out)[row1 * N + n1] = (bf16_t)v11;
    }
  }
}

// ---------------- V transpose: vT[b][h][d][s] ----------------
__global__ void vtrans_k(const bf16_t* __restrict__ qkv, bf16_t* __restrict__ vT) {
  int idx = blockIdx.x * 256 + threadIdx.x;
  int s = idx & 511;
  int d = (idx >> 9) & 63;
  int h = (idx >> 15) & 7;
  int b = idx >> 18;
  vT[idx] = qkv[((size_t)(b * SS + s)) * (3 * DD) + 2 * DD + h * HD + d];
}

// ---------------- fused flash attention ----------------
// grid: B*H*(S/64) blocks of 128 threads; wave = one 16-row q tile.
// Q was pre-scaled by 1/sqrt(HD) in the QKV GEMM epilogue.
__global__ void attention_k(const bf16_t* __restrict__ qkv,
                            const bf16_t* __restrict__ vT,
                            bf16_t* __restrict__ ctx) {
  int bid = blockIdx.x;
  int qc = bid & 7;
  int h  = (bid >> 3) & 7;
  int b  = bid >> 6;
  int wave = threadIdx.x >> 5;
  int lane = threadIdx.x & 31;
  int hf = lane >> 4;
  int lq = lane & 15;
  int m0 = qc * 64 + wave * 16;

  __shared__ __align__(16) unsigned char smem[4 * 16 * 80]; // P staging
  unsigned char* my = smem + wave * (16 * 80);

  const bf16_t* qrow = qkv + (size_t)(b * SS + m0 + lq) * (3 * DD) + h * HD;
  v16bf qa0 = load_a2(qrow + 16 * hf,      qrow + 8 + 16 * hf);
  v16bf qa1 = load_a2(qrow + 32 + 16 * hf, qrow + 40 + 16 * hf);

  const bf16_t* kbase = qkv + (size_t)(b * SS) * (3 * DD) + DD + h * HD;
  const bf16_t* vbase = vT + (size_t)((b * HH + h) * HD) * SS;

  float mrow[8], lrow[8];
  v8f o0 = {}, o1 = {}, o2 = {}, o3 = {};
#pragma unroll
  for (int r = 0; r < 8; ++r) { mrow[r] = -3.0e38f; lrow[r] = 0.f; }

  for (int j = 0; j < SS / 32; ++j) {
    const bf16_t* k0p = kbase + (size_t)(j * 32 + lq) * (3 * DD) + 16 * hf;
    const bf16_t* k1p = k0p + (size_t)16 * (3 * DD);
    v8f s0 = {}, s1 = {};
    s0 = bwmma(qa0, *(const v16bf*)(k0p),      s0);
    s0 = bwmma(qa1, *(const v16bf*)(k0p + 32), s0);
    s1 = bwmma(qa0, *(const v16bf*)(k1p),      s1);
    s1 = bwmma(qa1, *(const v16bf*)(k1p + 32), s1);

#pragma unroll
    for (int r = 0; r < 8; ++r) {
      float a0 = s0[r], a1 = s1[r];
      float mn = fmaxf(mrow[r], redmax16(fmaxf(a0, a1)));
      float al = __expf(mrow[r] - mn);
      float p0 = __expf(a0 - mn);
      float p1 = __expf(a1 - mn);
      lrow[r] = lrow[r] * al + redsum16(p0 + p1);
      mrow[r] = mn;
      o0[r] *= al; o1[r] *= al; o2[r] *= al; o3[r] *= al;
      bf16_t* prow = (bf16_t*)(my + (r + 8 * hf) * 80);
      prow[lq]      = (bf16_t)p0;
      prow[lq + 16] = (bf16_t)p1;
    }
    // reload P in A-fragment layout (same-wave DS ops are in order)
    const bf16_t* pr = (const bf16_t*)(my + lq * 80);
    v16bf pa = load_a2(pr + 16 * hf, pr + 16 * hf + 8);

    const bf16_t* vj = vbase + j * 32 + 16 * hf;
    o0 = bwmma(pa, *(const v16bf*)(vj + (size_t)(lq)      * SS), o0);
    o1 = bwmma(pa, *(const v16bf*)(vj + (size_t)(16 + lq) * SS), o1);
    o2 = bwmma(pa, *(const v16bf*)(vj + (size_t)(32 + lq) * SS), o2);
    o3 = bwmma(pa, *(const v16bf*)(vj + (size_t)(48 + lq) * SS), o3);
  }

#pragma unroll
  for (int r = 0; r < 8; ++r) {
    float inv = 1.f / lrow[r];
    size_t row = (size_t)(b * SS + m0 + r + 8 * hf);
    bf16_t* cr = ctx + row * DD + h * HD;
    cr[lq]      = (bf16_t)(o0[r] * inv);
    cr[16 + lq] = (bf16_t)(o1[r] * inv);
    cr[32 + lq] = (bf16_t)(o2[r] * inv);
    cr[48 + lq] = (bf16_t)(o3[r] * inv);
  }
}

// ---------------- input projection + positional encoding ----------------
__global__ void input_k(const float* __restrict__ feat, const int* __restrict__ pos,
                        const float* __restrict__ w_in, const float* __restrict__ b_in,
                        float* __restrict__ x, bf16_t* __restrict__ xb) {
  int t = blockIdx.x;
  int d = threadIdx.x;
  const float* f = feat + (size_t)t * INF_;
  float acc = b_in[d];
#pragma unroll
  for (int i = 0; i < INF_; ++i) acc += f[i] * w_in[i * DD + d];
  float p = (float)pos[t];
  int e = d & 63;
  int i2 = e >> 1;
  float freq = __expf(-(float)(2 * i2) * (9.2103403719761836f / 64.f));
  float ang = p * freq;
  acc += (e & 1) ? __cosf(ang) : __sinf(ang);
  x[(size_t)t * DD + d]  = acc;
  xb[(size_t)t * DD + d] = (bf16_t)acc;
}

// ---------------- residual add + LayerNorm (wave per row) ----------------
__global__ void addln_k(const float* __restrict__ y,
                        const float* __restrict__ g, const float* __restrict__ bta,
                        float* __restrict__ x, bf16_t* __restrict__ xb) {
  int wave = threadIdx.x >> 5;
  int lane = threadIdx.x & 31;
  size_t row = (size_t)(blockIdx.x * 8 + wave);
  const float* xr = x + row * DD;
  const float* yr = y + row * DD;
  float z[16];
  float s = 0.f;
#pragma unroll
  for (int i = 0; i < 16; ++i) { z[i] = xr[i * 32 + lane] + yr[i * 32 + lane]; s += z[i]; }
  s = redsum32(s);
  float mu = s * (1.f / DD);
  float v = 0.f;
#pragma unroll
  for (int i = 0; i < 16; ++i) { float d = z[i] - mu; v += d * d; }
  v = redsum32(v);
  float rs = rsqrtf(v * (1.f / DD) + 1e-5f);
  float* xo = x + row * DD;
  bf16_t* xbo = xb + row * DD;
#pragma unroll
  for (int i = 0; i < 16; ++i) {
    int c = i * 32 + lane;
    float nv = (z[i] - mu) * rs * g[c] + bta[c];
    xo[c]  = nv;
    xbo[c] = (bf16_t)nv;
  }
}

// ---------------- output projection D -> 1 (wave per token) ----------------
__global__ void outproj_k(const float* __restrict__ x, const float* __restrict__ w,
                          const float* __restrict__ b, float* __restrict__ out) {
  int wave = threadIdx.x >> 5;
  int lane = threadIdx.x & 31;
  size_t row = (size_t)(blockIdx.x * 8 + wave);
  const float* xr = x + row * DD;
  float s = 0.f;
#pragma unroll
  for (int i = 0; i < 16; ++i) s += xr[i * 32 + lane] * w[i * 32 + lane];
  s = redsum32(s);
  if (lane == 0) out[row] = s + b[0];
}

// ---------------- orchestration ----------------
extern "C" void kernel_launch(void* const* d_in, const int* in_sizes, int n_in,
                              void* d_out, int out_size, void* d_ws, size_t ws_size,
                              hipStream_t stream) {
  const float* features = (const float*)d_in[0];
  const int*   pos      = (const int*)d_in[1];
  const float* w_in     = (const float*)d_in[2];
  const float* b_in     = (const float*)d_in[3];
  const float* qkv_w    = (const float*)d_in[4];
  const float* qkv_b    = (const float*)d_in[5];
  const float* ow       = (const float*)d_in[6];
  const float* ob       = (const float*)d_in[7];
  const float* g1       = (const float*)d_in[8];
  const float* b1       = (const float*)d_in[9];
  const float* f1w      = (const float*)d_in[10];
  const float* f1b      = (const float*)d_in[11];
  const float* f2w      = (const float*)d_in[12];
  const float* f2b      = (const float*)d_in[13];
  const float* g2       = (const float*)d_in[14];
  const float* b2       = (const float*)d_in[15];
  const float* w_out    = (const float*)d_in[16];
  const float* b_out    = (const float*)d_in[17];
  float* out = (float*)d_out;

  char* ws = (char*)d_ws;
  float*  x   = (float*)(ws + OFF_X);
  bf16_t* xb  = (bf16_t*)(ws + OFF_XB);
  float*  y   = (float*)(ws + OFF_Y);
  bf16_t* qkv = (bf16_t*)(ws + OFF_QKV);
  bf16_t* vT  = (bf16_t*)(ws + OFF_VT);
  bf16_t* ctx = (bf16_t*)(ws + OFF_CTX);
  bf16_t* hbf = (bf16_t*)(ws + OFF_H);
  bf16_t* pw  = (bf16_t*)(ws + OFF_W);

  for (int l = 0; l < LL; ++l) {
    bf16_t* pl = pw + (size_t)l * PW_LAYER;
    pack_b_k<<<(int)((PW_QKV + 255) / 256), 256, 0, stream>>>(
        qkv_w + (size_t)l * PW_QKV, pl, DD, 3 * DD);
    pack_b_k<<<(int)((PW_OW + 255) / 256), 256, 0, stream>>>(
        ow + (size_t)l * PW_OW, pl + PW_QKV, DD, DD);
    pack_b_k<<<(int)((PW_F1 + 255) / 256), 256, 0, stream>>>(
        f1w + (size_t)l * PW_F1, pl + PW_QKV + PW_OW, DD, FF_);
    pack_b_k<<<(int)((PW_F2 + 255) / 256), 256, 0, stream>>>(
        f2w + (size_t)l * PW_F2, pl + PW_QKV + PW_OW + PW_F1, FF_, DD);
  }

  input_k<<<TOK, DD, 0, stream>>>(features, pos, w_in, b_in, x, xb);

  for (int l = 0; l < LL; ++l) {
    bf16_t* pl = pw + (size_t)l * PW_LAYER;
    // QKV projection -> bf16 (Q columns pre-scaled by 1/sqrt(HD))
    wmma_gemm_k<3><<<(TOK / 32) * (3 * DD / 32) / 4, 128, 0, stream>>>(
        xb, pl, qkv_b + (size_t)l * 3 * DD, qkv, TOK, 3 * DD, DD);
    vtrans_k<<<(BB * HH * HD * SS) / 256, 256, 0, stream>>>(qkv, vT);
    attention_k<<<BB * HH * (SS / 64), 128, 0, stream>>>(qkv, vT, ctx);
    wmma_gemm_k<0><<<(TOK / 32) * (DD / 32) / 4, 128, 0, stream>>>(
        ctx, pl + PW_QKV, ob + (size_t)l * DD, y, TOK, DD, DD);
    addln_k<<<TOK / 8, 256, 0, stream>>>(y, g1 + (size_t)l * DD, b1 + (size_t)l * DD, x, xb);
    wmma_gemm_k<2><<<(TOK / 32) * (FF_ / 32) / 4, 128, 0, stream>>>(
        xb, pl + PW_QKV + PW_OW, f1b + (size_t)l * FF_, hbf, TOK, FF_, DD);
    wmma_gemm_k<0><<<(TOK / 32) * (DD / 32) / 4, 128, 0, stream>>>(
        hbf, pl + PW_QKV + PW_OW + PW_F1, f2b + (size_t)l * DD, y, TOK, DD, FF_);
    addln_k<<<TOK / 8, 256, 0, stream>>>(y, g2 + (size_t)l * DD, b2 + (size_t)l * DD, x, xb);
  }

  outproj_k<<<TOK / 8, 256, 0, stream>>>(x, w_out, b_out, out);
}
